// SelectiveRWKVBlock_28183575396838
// MI455X (gfx1250) — matrix operand order repaired and checked by reading, not compiled, not run on hardware
//
#include <hip/hip_runtime.h>
#include <hip/hip_bf16.h>

// ---------------------------------------------------------------------------
// SelectiveRWKVBlock for MI455X (gfx1250, wave32, WMMA)
//   B=2 L=2048 D=2560 H=40 HD=64 S=64 K=4
//   GEMMs (~480 GFLOP) -> v_wmma_f32_16x16x32_bf16, f32 accumulate.
//   Weights converted to bf16 once per launch (~118MB, fits in 192MB L2).
//   Scan: 80 blocks (B*H), state in registers, per-step operands double-
//   buffered in LDS via async global->LDS copies (ASYNCcnt) when available.
// ---------------------------------------------------------------------------

typedef __attribute__((ext_vector_type(16))) __bf16 v16bf;
typedef __attribute__((ext_vector_type(8)))  __bf16 v8bf;
typedef __attribute__((ext_vector_type(8)))  float  v8f;

#define DIMC  2560
#define HEADS 40
#define HDIM  64
#define SDIM  64
#define KCONV 4
#define N3    7680   // H*3*S
#define LSEQ  2048
#define BATCH 2
#define MROWS (BATCH*LSEQ)   // 4096

// ---- CDNA5 async global->LDS support probes -------------------------------
#if defined(__has_builtin)
#if __has_builtin(__builtin_amdgcn_global_load_async_to_lds_b32)
#define HAVE_ASYNC_LDS 1
#endif
#if __has_builtin(__builtin_amdgcn_s_wait_asynccnt)
#define WAIT_ASYNC(n) __builtin_amdgcn_s_wait_asynccnt(n)
#endif
#endif
#ifndef WAIT_ASYNC
#define WAIT_ASYNC(n) asm volatile("s_wait_asynccnt %0" ::"i"(n))
#endif

#ifdef HAVE_ASYNC_LDS
// per-lane async copy of 4 bytes: LDS[dst] = MEM[src]; tracked by ASYNCcnt.
// Builtin signature (per hipcc diagnostic): (global int*, shared int*, imm, cpol)
#define ASYNC_CP_B32(ldsdst, gsrc)                                          \
    __builtin_amdgcn_global_load_async_to_lds_b32(                          \
        (__attribute__((address_space(1))) int*)(gsrc),                     \
        (__attribute__((address_space(3))) int*)(ldsdst), 0, 0)
#endif

// ------------------------------- cast f32 -> bf16 --------------------------
__global__ void cast_f32_bf16_kernel(const float* __restrict__ in,
                                     __bf16* __restrict__ out, size_t n) {
    size_t i = (size_t)blockIdx.x * blockDim.x + threadIdx.x;
    if (i < n) out[i] = (__bf16)in[i];
}

// ----------------------- causal depthwise conv + residual ------------------
__global__ void conv_residual_kernel(const float* __restrict__ x,
                                     const float* __restrict__ cw,
                                     float* __restrict__ xsum) {
    size_t i = (size_t)blockIdx.x * blockDim.x + threadIdx.x;
    size_t total = (size_t)MROWS * DIMC;
    if (i >= total) return;
    int d = (int)(i % DIMC);
    size_t bl = i / DIMC;
    int l = (int)(bl % LSEQ);
    float s = x[i];
    #pragma unroll
    for (int j = 0; j < KCONV; ++j) {
        int ls = l - (KCONV - 1) + j;
        if (ls >= 0) {
            size_t idx = i - (size_t)(l - ls) * DIMC;
            s += cw[d * KCONV + j] * x[idx];
        }
    }
    xsum[i] = s;
}

// ------------------------------- LayerNorm ---------------------------------
__global__ __launch_bounds__(256) void layernorm_kernel(
    const float* __restrict__ a, const float* __restrict__ b,
    const float* __restrict__ g, const float* __restrict__ beta,
    float* __restrict__ outf, __bf16* __restrict__ outbf) {
    const int row = blockIdx.x;
    const int tid = threadIdx.x;
    const size_t base = (size_t)row * DIMC;
    const int PER = DIMC / 256;  // 10
    float vals[PER];
    float s = 0.f, ss = 0.f;
    #pragma unroll
    for (int i = 0; i < PER; ++i) {
        int c = i * 256 + tid;
        float v = a[base + c];
        if (b) v += b[base + c];
        vals[i] = v;
        s += v; ss += v * v;
    }
    #pragma unroll
    for (int o = 16; o > 0; o >>= 1) {
        s  += __shfl_xor(s,  o, 32);
        ss += __shfl_xor(ss, o, 32);
    }
    __shared__ float red_s[8], red_ss[8];
    if ((tid & 31) == 0) { red_s[tid >> 5] = s; red_ss[tid >> 5] = ss; }
    __syncthreads();
    float ts = 0.f, tss = 0.f;
    #pragma unroll
    for (int w = 0; w < 8; ++w) { ts += red_s[w]; tss += red_ss[w]; }
    const float inv = 1.0f / (float)DIMC;
    float mean = ts * inv;
    float var  = tss * inv - mean * mean;
    float rsq  = rsqrtf(var + 1e-5f);
    #pragma unroll
    for (int i = 0; i < PER; ++i) {
        int c = i * 256 + tid;
        float o = (vals[i] - mean) * rsq * g[c] + beta[c];
        outf[base + c] = o;
        if (outbf) outbf[base + c] = (__bf16)o;
    }
}

// --------------------------- WMMA bf16 GEMM --------------------------------
// C[M,N] (f32) = A[M,K] (bf16, row-major) x W[N,K]^T (bf16, row-major)
// Block = 128 threads = 4 waves (2x2), wave = 64x64 via 4x4 WMMA tiles.
__device__ __forceinline__ v16bf load_frag_a(const __bf16* __restrict__ A,
                                             int ld, int row0, int k0, int lane) {
    const __bf16* p = A + (size_t)(row0 + (lane & 15)) * ld + k0 + ((lane >> 4) << 3);
    v8bf lo = *(const v8bf*)(p);
    v8bf hi = *(const v8bf*)(p + 16);
    v16bf t;
    #pragma unroll
    for (int e = 0; e < 8; ++e) { t[e] = lo[e]; t[8 + e] = hi[e]; }
    return t;
}
__device__ __forceinline__ v16bf load_frag_b(const __bf16* __restrict__ W,
                                             int ld, int col0, int k0, int lane) {
    const __bf16* p = W + (size_t)(col0 + (lane & 15)) * ld + k0 + ((lane >> 4) << 3);
    v8bf lo = *(const v8bf*)(p);
    v8bf hi = *(const v8bf*)(p + 16);
    v16bf t;
    #pragma unroll
    for (int e = 0; e < 8; ++e) { t[e] = lo[e]; t[8 + e] = hi[e]; }
    return t;
}

__global__ __launch_bounds__(128) void gemm_bf16_wmma_kernel(
    const __bf16* __restrict__ A, const __bf16* __restrict__ W,
    float* __restrict__ C, int M, int N, int Kd) {
    const int lane = threadIdx.x & 31;
    const int wave = threadIdx.x >> 5;     // 0..3
    const int wm = wave >> 1, wn = wave & 1;
    const int m0 = blockIdx.y * 128 + wm * 64;
    const int n0 = blockIdx.x * 128 + wn * 64;

    v8f acc[4][4] = {};
    for (int k0 = 0; k0 < Kd; k0 += 32) {
        v16bf af[4], bfm[4];
        #pragma unroll
        for (int i = 0; i < 4; ++i) af[i]  = load_frag_a(A, Kd, m0 + i * 16, k0, lane);
        #pragma unroll
        for (int j = 0; j < 4; ++j) bfm[j] = load_frag_b(W, Kd, n0 + j * 16, k0, lane);
        if (k0 + 32 < Kd) {
            __builtin_prefetch(A + (size_t)(m0 + (lane & 15)) * Kd + k0 + 32, 0, 1);
            __builtin_prefetch(W + (size_t)(n0 + (lane & 15)) * Kd + k0 + 32, 0, 1);
        }
        #pragma unroll
        for (int i = 0; i < 4; ++i)
            #pragma unroll
            for (int j = 0; j < 4; ++j)
                acc[i][j] = __builtin_amdgcn_wmma_f32_16x16x32_bf16(
                    false, af[i], false, bfm[j], (short)0, acc[i][j], false, false);
    }
    const int rb  = (lane >> 4) << 3;
    const int col = lane & 15;
    #pragma unroll
    for (int i = 0; i < 4; ++i)
        #pragma unroll
        for (int j = 0; j < 4; ++j)
            #pragma unroll
            for (int e = 0; e < 8; ++e)
                C[(size_t)(m0 + i * 16 + rb + e) * N + (n0 + j * 16 + col)] = acc[i][j][e];
}

// -------------------- gather dt_raw out of proj, cast bf16 ------------------
__global__ void extract_dtraw_kernel(const float* __restrict__ proj,
                                     __bf16* __restrict__ out) {
    size_t i = (size_t)blockIdx.x * blockDim.x + threadIdx.x;
    size_t total = (size_t)MROWS * DIMC;
    if (i >= total) return;
    int c = (int)(i % DIMC);
    size_t bl = i / DIMC;
    int h = c >> 6, s = c & 63;
    out[i] = (__bf16)proj[bl * N3 + h * 192 + s];
}

// -------- fused elementwise: sigmoid(r), sigmoid(g), kv=k*v, w=sig(-dt) -----
__global__ void epilogue1_kernel(float* __restrict__ r, float* __restrict__ k,
                                 const float* __restrict__ v, float* __restrict__ g,
                                 float* __restrict__ dt, const float* __restrict__ bdt) {
    size_t i = (size_t)blockIdx.x * blockDim.x + threadIdx.x;
    size_t total = (size_t)MROWS * DIMC;
    if (i >= total) return;
    int c = (int)(i % DIMC);
    r[i] = 1.f / (1.f + __expf(-r[i]));
    g[i] = 1.f / (1.f + __expf(-g[i]));
    k[i] = k[i] * v[i];
    dt[i] = 1.f / (1.f + __expf(dt[i] + bdt[c]));  // exp(-softplus) == sigmoid(-t)
}

// ------------------------------ recurrent scan ------------------------------
// One block per (b,h). state[64 s][64 d] in registers: thread t owns
// d = t&63 and s-chunk [(t>>6)*16, +16). Per-step operands (w,B,C,kv; 1KB)
// double-buffered in LDS, filled by async global->LDS copies one step ahead
// (s_wait_asynccnt 4 => only the older step's loads must be complete).
__global__ __launch_bounds__(256) void scan_kernel(
    const float* __restrict__ proj, const float* __restrict__ wdec,
    const float* __restrict__ kv, const float* __restrict__ r,
    const float* __restrict__ g, __bf16* __restrict__ attn_bf,
    float* __restrict__ final_state) {
    const int bh = blockIdx.x;             // 0..B*H-1
    const int b = bh / HEADS, h = bh % HEADS;
    const int t = threadIdx.x;
    const int d = t & 63;
    const int s0 = (t >> 6) << 4;          // 0,16,32,48
    float st[16];
    #pragma unroll
    for (int i = 0; i < 16; ++i) st[i] = 0.f;

    __shared__ float stage[2][4][64];      // [buf][w,B,C,kv][64]
    __shared__ float sy[64];

    const size_t baseD = (size_t)b * LSEQ * DIMC + (size_t)h * 64;
    const size_t baseP = (size_t)b * LSEQ * N3 + (size_t)h * 192;

#ifdef HAVE_ASYNC_LDS
    // prologue: issue async fill of buffer 0 with step 0 operands
    if (t < 64) {
        size_t pd = baseD + t;
        size_t pp = baseP + t;
        ASYNC_CP_B32(&stage[0][0][t], wdec + pd);
        ASYNC_CP_B32(&stage[0][1][t], proj + pp + 64);
        ASYNC_CP_B32(&stage[0][2][t], proj + pp + 128);
        ASYNC_CP_B32(&stage[0][3][t], kv + pd);
    }
#else
    float rw = 0.f, rB = 0.f, rC = 0.f, rkv = 0.f;
    if (t < 64) {
        size_t pd = baseD + t;
        size_t pp = baseP + t;
        rw  = wdec[pd];
        rB  = proj[pp + 64];
        rC  = proj[pp + 128];
        rkv = kv[pd];
    }
#endif

    for (int l = 0; l < LSEQ; ++l) {
        const int cur = l & 1;
        if (t < 64) sy[t] = 0.f;
#ifdef HAVE_ASYNC_LDS
        if (l + 1 < LSEQ) {
            if (t < 64) {  // prefetch step l+1 into the other buffer
                size_t pd = baseD + (size_t)(l + 1) * DIMC + t;
                size_t pp = baseP + (size_t)(l + 1) * N3 + t;
                ASYNC_CP_B32(&stage[cur ^ 1][0][t], wdec + pd);
                ASYNC_CP_B32(&stage[cur ^ 1][1][t], proj + pp + 64);
                ASYNC_CP_B32(&stage[cur ^ 1][2][t], proj + pp + 128);
                ASYNC_CP_B32(&stage[cur ^ 1][3][t], kv + pd);
            }
            WAIT_ASYNC(4);   // newest 4 (step l+1) may fly; step l's are done
        } else {
            WAIT_ASYNC(0);
        }
#else
        if (t < 64) {        // publish registers for step l
            stage[cur][0][t] = rw;
            stage[cur][1][t] = rB;
            stage[cur][2][t] = rC;
            stage[cur][3][t] = rkv;
        }
#endif
        __syncthreads();
#ifndef HAVE_ASYNC_LDS
        if (t < 64 && l + 1 < LSEQ) {  // overlap next-step loads with compute
            size_t pd = baseD + (size_t)(l + 1) * DIMC + t;
            size_t pp = baseP + (size_t)(l + 1) * N3 + t;
            rw  = wdec[pd];
            rB  = proj[pp + 64];
            rC  = proj[pp + 128];
            rkv = kv[pd];
        }
#endif
        const float* sw  = stage[cur][0];
        const float* sB  = stage[cur][1];
        const float* sC  = stage[cur][2];
        const float  kvv = stage[cur][3][d];
        float acc = 0.f;
        #pragma unroll
        for (int i = 0; i < 16; ++i) {
            int s = s0 + i;
            float ns = st[i] * sw[s] + sB[s] * kvv;
            st[i] = ns;
            acc += sC[s] * ns;
        }
        atomicAdd(&sy[d], acc);            // ds_add_f32, 4-way per address
        __syncthreads();
        if (t < 64) {
            size_t pd = baseD + (size_t)l * DIMC + t;
            attn_bf[pd] = (__bf16)(r[pd] * sy[t] * g[pd]);
        }
    }
    const size_t fs = (size_t)bh * SDIM * HDIM;
    #pragma unroll
    for (int i = 0; i < 16; ++i)
        final_state[fs + (size_t)(s0 + i) * HDIM + d] = st[i];
}

// ---------------------------------------------------------------------------
extern "C" void kernel_launch(void* const* d_in, const int* in_sizes, int n_in,
                              void* d_out, int out_size, void* d_ws, size_t ws_size,
                              hipStream_t stream) {
    const float* x      = (const float*)d_in[0];
    const float* W_r    = (const float*)d_in[1];
    const float* W_k    = (const float*)d_in[2];
    const float* W_v    = (const float*)d_in[3];
    const float* W_g    = (const float*)d_in[4];
    const float* W_xp   = (const float*)d_in[5];
    const float* W_dt   = (const float*)d_in[6];
    const float* b_dt   = (const float*)d_in[7];
    const float* conv_w = (const float*)d_in[8];
    const float* ln1_g  = (const float*)d_in[9];
    const float* ln1_b  = (const float*)d_in[10];
    const float* ln2_g  = (const float*)d_in[11];
    const float* ln2_b  = (const float*)d_in[12];
    const float* W_out  = (const float*)d_in[13];
    float* out = (float*)d_out;

    const size_t MD = (size_t)MROWS * DIMC;
    const size_t DD = (size_t)DIMC * DIMC;
    const size_t XD = (size_t)N3 * DIMC;

    char* p = (char*)d_ws;
    auto alloc = [&](size_t bytes) -> void* {
        void* r = (void*)p;
        p += (bytes + 255) & ~(size_t)255;
        return r;
    };
    float*  xsum   = (float*)alloc(MD * 4);
    float*  xn     = (float*)alloc(MD * 4);
    __bf16* xn_bf  = (__bf16*)alloc(MD * 2);
    __bf16* Wr_bf  = (__bf16*)alloc(DD * 2);
    __bf16* Wk_bf  = (__bf16*)alloc(DD * 2);
    __bf16* Wv_bf  = (__bf16*)alloc(DD * 2);
    __bf16* Wg_bf  = (__bf16*)alloc(DD * 2);
    __bf16* Wo_bf  = (__bf16*)alloc(DD * 2);
    __bf16* Wdt_bf = (__bf16*)alloc(DD * 2);
    __bf16* Wxp_bf = (__bf16*)alloc(XD * 2);
    float*  proj   = (float*)alloc((size_t)MROWS * N3 * 4);
    __bf16* dtr_bf = (__bf16*)alloc(MD * 2);
    float*  rlin   = (float*)alloc(MD * 4);
    float*  klin   = (float*)alloc(MD * 4);   // becomes kv
    float*  vlin   = (float*)alloc(MD * 4);
    float*  glin   = (float*)alloc(MD * 4);
    float*  dtlin  = (float*)alloc(MD * 4);   // becomes decay w
    __bf16* attn_bf= (__bf16*)alloc(MD * 2);
    float*  woutres= (float*)alloc(MD * 4);
    (void)ws_size; (void)n_in; (void)in_sizes; (void)out_size;

    auto cgrid = [](size_t n) { return dim3((unsigned)((n + 255) / 256)); };

    cast_f32_bf16_kernel<<<cgrid(DD), 256, 0, stream>>>(W_r,  Wr_bf,  DD);
    cast_f32_bf16_kernel<<<cgrid(DD), 256, 0, stream>>>(W_k,  Wk_bf,  DD);
    cast_f32_bf16_kernel<<<cgrid(DD), 256, 0, stream>>>(W_v,  Wv_bf,  DD);
    cast_f32_bf16_kernel<<<cgrid(DD), 256, 0, stream>>>(W_g,  Wg_bf,  DD);
    cast_f32_bf16_kernel<<<cgrid(DD), 256, 0, stream>>>(W_out,Wo_bf,  DD);
    cast_f32_bf16_kernel<<<cgrid(DD), 256, 0, stream>>>(W_dt, Wdt_bf, DD);
    cast_f32_bf16_kernel<<<cgrid(XD), 256, 0, stream>>>(W_xp, Wxp_bf, XD);

    conv_residual_kernel<<<cgrid(MD), 256, 0, stream>>>(x, conv_w, xsum);
    layernorm_kernel<<<dim3(MROWS), 256, 0, stream>>>(xsum, nullptr, ln1_g, ln1_b,
                                                      xn, xn_bf);

    gemm_bf16_wmma_kernel<<<dim3(N3 / 128, MROWS / 128), 128, 0, stream>>>(
        xn_bf, Wxp_bf, proj, MROWS, N3, DIMC);

    dim3 gsq(DIMC / 128, MROWS / 128);
    gemm_bf16_wmma_kernel<<<gsq, 128, 0, stream>>>(xn_bf, Wr_bf, rlin, MROWS, DIMC, DIMC);
    gemm_bf16_wmma_kernel<<<gsq, 128, 0, stream>>>(xn_bf, Wk_bf, klin, MROWS, DIMC, DIMC);
    gemm_bf16_wmma_kernel<<<gsq, 128, 0, stream>>>(xn_bf, Wv_bf, vlin, MROWS, DIMC, DIMC);
    gemm_bf16_wmma_kernel<<<gsq, 128, 0, stream>>>(xn_bf, Wg_bf, glin, MROWS, DIMC, DIMC);

    extract_dtraw_kernel<<<cgrid(MD), 256, 0, stream>>>(proj, dtr_bf);
    gemm_bf16_wmma_kernel<<<gsq, 128, 0, stream>>>(dtr_bf, Wdt_bf, dtlin, MROWS, DIMC, DIMC);

    epilogue1_kernel<<<cgrid(MD), 256, 0, stream>>>(rlin, klin, vlin, glin, dtlin, b_dt);

    scan_kernel<<<dim3(BATCH * HEADS), 256, 0, stream>>>(
        proj, dtlin, klin, rlin, glin, attn_bf, out + MD);

    gemm_bf16_wmma_kernel<<<gsq, 128, 0, stream>>>(attn_bf, Wo_bf, woutres, MROWS, DIMC, DIMC);
    layernorm_kernel<<<dim3(MROWS), 256, 0, stream>>>(woutres, xn, ln2_g, ln2_b,
                                                      out, nullptr);
}
